// CNNCrossPatchBackbone_16209206575182
// MI455X (gfx1250) — compile-verified, matrix operand order checked by reading, not run on hardware
//
#include <hip/hip_runtime.h>
#include <hip/hip_bf16.h>
#include <hip/hip_fp16.h>

typedef _Float16 v2h  __attribute__((ext_vector_type(2)));
typedef _Float16 v8h  __attribute__((ext_vector_type(8)));
typedef _Float16 v16h __attribute__((ext_vector_type(16)));
typedef float    v8f  __attribute__((ext_vector_type(8)));

#define BATCH 4
#define KTOK  2048
#define DIM   1024
#define NHEAD 16
#define NCTX  1024
#define HDIM  64
#define KT    64   // GEMM K-tile depth (halves)

// 16-bit A/B fragment K-index for pair j (0..7), half-group hi (0/1):
// VGPR j<4: K = hi*8 + 2j ; VGPR j>=4: K = 16 + hi*8 + 2(j-4)   (ISA 7.12.2)
__device__ __forceinline__ int kfrag(int j, int hi) {
    return ((j & 4) << 2) + hi * 8 + ((j & 3) << 1);
}

// CDNA5 async VMEM->LDS copy, 16B per lane, tracked by ASYNCcnt.
// Flat->LDS truncation rule: low 32 bits of a generic shared pointer are the
// LDS byte address, which is what VDST carries.
__device__ __forceinline__ void async_b128(uint32_t lds_addr, const void* gptr) {
    asm volatile("global_load_async_to_lds_b128 %0, %1, off"
                 :: "v"(lds_addr), "v"((unsigned long long)(size_t)gptr)
                 : "memory");
}
__device__ __forceinline__ void wait_async0() {
    asm volatile("s_wait_asynccnt 0" ::: "memory");
}

// ---------------------------------------------------------------------------
// 1) Stable partition: ctx indices first, then tgt indices (per batch).
// ---------------------------------------------------------------------------
__global__ void partition_kernel(const unsigned char* __restrict__ is_ctx,
                                 int* __restrict__ order) {
    if (threadIdx.x != 0) return;
    int b = blockIdx.x;
    const unsigned char* src = is_ctx + (size_t)b * KTOK;
    int* dst = order + (size_t)b * KTOK;
    int c = 0, t = 0;
    for (int i = 0; i < KTOK; ++i) {
        if (src[i]) dst[c++] = i;
        else        dst[NCTX + (t++)] = i;
    }
}

// ---------------------------------------------------------------------------
// 2) fp32 -> f16 weight conversion (grid-stride).
// ---------------------------------------------------------------------------
__global__ void f32_to_f16_kernel(const float* __restrict__ src,
                                  _Float16* __restrict__ dst, int n) {
    for (int i = blockIdx.x * blockDim.x + threadIdx.x; i < n;
         i += gridDim.x * blockDim.x)
        dst[i] = (_Float16)src[i];
}

// ---------------------------------------------------------------------------
// 3) 2D RoPE + gather into ordered token buffer (f16).
// ---------------------------------------------------------------------------
__global__ void rope_gather_kernel(const float* __restrict__ x,
                                   const float* __restrict__ coords,
                                   const float* __restrict__ cache,
                                   const int* __restrict__ order,
                                   _Float16* __restrict__ tok) {
    int bp = blockIdx.x;
    int b = bp >> 11;            // KTOK == 2048
    int p = bp & (KTOK - 1);
    int i = order[(size_t)b * KTOK + p];
    float cy = coords[((size_t)b * KTOK + i) * 2 + 0];
    float cx = coords[((size_t)b * KTOK + i) * 2 + 1];
    const float sclp = 1023.0f / 224.0f;   // (MAX_POS-1)/IMAGE_SIZE
    float fy = fminf(fmaxf(cy * sclp, 0.0f), 1023.0f);
    float fx = fminf(fmaxf(cx * sclp, 0.0f), 1023.0f);
    int yp = (int)fy, xp = (int)fx;
    const float* src = x + ((size_t)b * KTOK + i) * DIM;
    _Float16* dst = tok + ((size_t)b * KTOK + p) * DIM;
    for (int j = threadIdx.x; j < 256; j += blockDim.x) {
        float c  = cache[((size_t)xp * 256 + j) * 2 + 0];
        float s  = cache[((size_t)xp * 256 + j) * 2 + 1];
        float p0 = src[2 * j], p1 = src[2 * j + 1];
        dst[2 * j]     = (_Float16)(p0 * c - p1 * s);
        dst[2 * j + 1] = (_Float16)(p0 * s + p1 * c);
        float c2 = cache[((size_t)yp * 256 + j) * 2 + 0];
        float s2 = cache[((size_t)yp * 256 + j) * 2 + 1];
        float q0 = src[512 + 2 * j], q1 = src[512 + 2 * j + 1];
        dst[512 + 2 * j]     = (_Float16)(q0 * c2 - q1 * s2);
        dst[512 + 2 * j + 1] = (_Float16)(q0 * s2 + q1 * c2);
    }
}

// ---------------------------------------------------------------------------
// 4) QKV projection GEMM:  Y = A @ W^T + bias.  Block 256 thr = 8 waves,
//    block tile 128(M) x 64(N), per-wave 32x32 (4 accumulators), KT=64,
//    double-buffered async LDS staging. Output in head layout f16.
// ---------------------------------------------------------------------------
__global__ __launch_bounds__(256) void proj_qkv_kernel(
    const _Float16* __restrict__ A, int row_off,
    const _Float16* __restrict__ W, const float* __restrict__ bias, int col_off,
    _Float16* __restrict__ qb, _Float16* __restrict__ kb,
    _Float16* __restrict__ vb) {
    __shared__ _Float16 sA[2][128][KT];
    __shared__ _Float16 sW[2][64][KT];
    const int tid = threadIdx.x;
    const int wave = tid >> 5, lane = tid & 31;
    const int n = lane & 15, hi = lane >> 4;
    const int wm = wave >> 1, wn = wave & 1;
    const int n0 = blockIdx.x * 64;
    const int m0 = blockIdx.y * 128;

    auto stage = [&](int kt, int bf) {
        // A: 128x64 halves = 1024 b128 chunks -> 4 per thread
        for (int it = 0; it < 4; ++it) {
            int idx = tid + it * 256;
            int r = idx >> 3, c8 = idx & 7;
            int g = m0 + r, b = g >> 10, m = g & 1023;
            const _Float16* src =
                A + ((size_t)(b * KTOK + row_off + m)) * DIM + kt + 8 * c8;
            async_b128((uint32_t)(size_t)&sA[bf][r][8 * c8], src);
        }
        // W: 64x64 halves = 512 b128 chunks -> 2 per thread
        for (int it = 0; it < 2; ++it) {
            int idx = tid + it * 256;
            int r = idx >> 3, c8 = idx & 7;
            const _Float16* src =
                W + (size_t)(col_off + n0 + r) * DIM + kt + 8 * c8;
            async_b128((uint32_t)(size_t)&sW[bf][r][8 * c8], src);
        }
    };

    v8f c00, c01, c10, c11;
    for (int r = 0; r < 8; ++r) { c00[r] = c01[r] = c10[r] = c11[r] = 0.0f; }

    stage(0, 0);
    int bf = 0;
    for (int kt = 0; kt < DIM; kt += KT) {
        wait_async0();          // this tile's copies have landed (per wave)
        __syncthreads();        // ... for all waves
        if (kt + KT < DIM) stage(kt + KT, bf ^ 1);   // overlap with compute
        for (int ks = 0; ks < KT; ks += 32) {
            v16h a0, a1, b0, b1;
            for (int j = 0; j < 8; ++j) {
                int kf = ks + kfrag(j, hi);
                v2h t;
                t = *(const v2h*)&sA[bf][wm * 32 + n][kf];
                a0[2 * j] = t[0]; a0[2 * j + 1] = t[1];
                t = *(const v2h*)&sA[bf][wm * 32 + 16 + n][kf];
                a1[2 * j] = t[0]; a1[2 * j + 1] = t[1];
                t = *(const v2h*)&sW[bf][wn * 32 + n][kf];
                b0[2 * j] = t[0]; b0[2 * j + 1] = t[1];
                t = *(const v2h*)&sW[bf][wn * 32 + 16 + n][kf];
                b1[2 * j] = t[0]; b1[2 * j + 1] = t[1];
            }
            c00 = __builtin_amdgcn_wmma_f32_16x16x32_f16(false, a0, false, b0,
                                                         (short)0, c00, false, false);
            c01 = __builtin_amdgcn_wmma_f32_16x16x32_f16(false, a0, false, b1,
                                                         (short)0, c01, false, false);
            c10 = __builtin_amdgcn_wmma_f32_16x16x32_f16(false, a1, false, b0,
                                                         (short)0, c10, false, false);
            c11 = __builtin_amdgcn_wmma_f32_16x16x32_f16(false, a1, false, b1,
                                                         (short)0, c11, false, false);
        }
        __syncthreads();        // reads done before this buffer is re-staged
        bf ^= 1;
    }

    for (int rb = 0; rb < 2; ++rb) {
        for (int cb = 0; cb < 2; ++cb) {
            const v8f& c = rb ? (cb ? c11 : c10) : (cb ? c01 : c00);
            for (int r = 0; r < 8; ++r) {
                int mloc = wm * 32 + rb * 16 + r + 8 * hi;
                int g = m0 + mloc, b = g >> 10, p = g & 1023;
                int n_eff = col_off + n0 + wn * 32 + cb * 16 + n;
                float val = c[r] + bias[n_eff];
                int sel = n_eff >> 10, dl = n_eff & 1023;
                int h = dl >> 6, hd = dl & 63;
                _Float16* dst = (sel == 0) ? qb : ((sel == 1) ? kb : vb);
                dst[(((size_t)b * NHEAD + h) * NCTX + p) * HDIM + hd] =
                    (_Float16)val;
            }
        }
    }
}

// ---------------------------------------------------------------------------
// 5) Flash-style attention. Block 128 thr = 4 waves; (b, h, 64 q rows);
//    per-wave 16 q rows x 64 head dims, online softmax over 1024 kv rows.
//    K staged via async-to-LDS; V staged transposed (manual).
// ---------------------------------------------------------------------------
__global__ __launch_bounds__(128) void attn_kernel(
    const _Float16* __restrict__ Q, const _Float16* __restrict__ Kb,
    const _Float16* __restrict__ Vb, _Float16* __restrict__ O, int row_off) {
    __shared__ _Float16 sK[32][64];
    __shared__ _Float16 sVt[64][32];   // V transposed: [head_dim][kv_row]
    __shared__ _Float16 sP[4][16][32]; // per-wave P tile (C-layout -> A-layout)
    const int tid = threadIdx.x;
    const int wave = tid >> 5, lane = tid & 31;
    const int n = lane & 15, hi = lane >> 4;
    const int h = blockIdx.y, b = blockIdx.z;
    const int q0 = blockIdx.x * 64 + wave * 16;
    const size_t head = ((size_t)b * NHEAD + h) * NCTX;

    // Q fragments (K = head dims 0..31 and 32..63), loaded once
    v16h aq0, aq1;
    {
        const _Float16* qrow = Q + (head + q0 + n) * HDIM;
        for (int j = 0; j < 8; ++j) {
            int kf = kfrag(j, hi);
            v2h t0 = *(const v2h*)(qrow + kf);
            v2h t1 = *(const v2h*)(qrow + 32 + kf);
            aq0[2 * j] = t0[0]; aq0[2 * j + 1] = t0[1];
            aq1[2 * j] = t1[0]; aq1[2 * j + 1] = t1[1];
        }
    }
    v8f o0, o1, o2, o3;
    float rmax[8], rsum[8];
    for (int r = 0; r < 8; ++r) {
        o0[r] = o1[r] = o2[r] = o3[r] = 0.0f;
        rmax[r] = -3.0e38f; rsum[r] = 0.0f;
    }
    const float scale = 0.125f;  // 1/sqrt(64)

    for (int kk = 0; kk < NCTX; kk += 32) {
        __syncthreads();
        // stage K (async b128) and V (b128 load + transposed stores)
        for (int it = 0; it < 2; ++it) {
            int idx = tid + it * 128;          // 256 chunks of 8 halves
            int r = idx >> 3, c8 = idx & 7;
            async_b128((uint32_t)(size_t)&sK[r][8 * c8],
                       Kb + (head + kk + r) * HDIM + 8 * c8);
            v8h tv = *(const v8h*)(Vb + (head + kk + r) * HDIM + 8 * c8);
            for (int e = 0; e < 8; ++e) sVt[8 * c8 + e][r] = tv[e];
        }
        wait_async0();
        __syncthreads();
        // S = Q @ K^T for 32 kv columns (two 16x16 tiles)
        v16h b00, b01, b10, b11;
        for (int j = 0; j < 8; ++j) {
            int kf = kfrag(j, hi);
            v2h t00 = *(const v2h*)&sK[n][kf];
            v2h t01 = *(const v2h*)&sK[n][32 + kf];
            v2h t10 = *(const v2h*)&sK[16 + n][kf];
            v2h t11 = *(const v2h*)&sK[16 + n][32 + kf];
            b00[2 * j] = t00[0]; b00[2 * j + 1] = t00[1];
            b01[2 * j] = t01[0]; b01[2 * j + 1] = t01[1];
            b10[2 * j] = t10[0]; b10[2 * j + 1] = t10[1];
            b11[2 * j] = t11[0]; b11[2 * j + 1] = t11[1];
        }
        v8f zs; for (int r = 0; r < 8; ++r) zs[r] = 0.0f;
        v8f c0 = __builtin_amdgcn_wmma_f32_16x16x32_f16(false, aq0, false, b00,
                                                        (short)0, zs, false, false);
        c0 = __builtin_amdgcn_wmma_f32_16x16x32_f16(false, aq1, false, b01,
                                                    (short)0, c0, false, false);
        v8f c1 = __builtin_amdgcn_wmma_f32_16x16x32_f16(false, aq0, false, b10,
                                                        (short)0, zs, false, false);
        c1 = __builtin_amdgcn_wmma_f32_16x16x32_f16(false, aq1, false, b11,
                                                    (short)0, c1, false, false);
        // online softmax; row stats identical across each 16-lane half-group
        for (int r = 0; r < 8; ++r) {
            float s0 = c0[r] * scale, s1 = c1[r] * scale;
            float v = fmaxf(s0, s1);
            for (int msk = 8; msk >= 1; msk >>= 1)
                v = fmaxf(v, __shfl_xor(v, msk, 32));
            float nm = fmaxf(rmax[r], v);
            float al = __expf(rmax[r] - nm);
            rmax[r] = nm;
            float p0 = __expf(s0 - nm), p1 = __expf(s1 - nm);
            float ps = p0 + p1;
            for (int msk = 8; msk >= 1; msk >>= 1)
                ps += __shfl_xor(ps, msk, 32);
            rsum[r] = rsum[r] * al + ps;
            o0[r] *= al; o1[r] *= al; o2[r] *= al; o3[r] *= al;
            int m = r + 8 * hi;
            sP[wave][m][n]      = (_Float16)p0;
            sP[wave][m][16 + n] = (_Float16)p1;
        }
        __syncthreads();
        // O += P @ V : one A fragment (16x32) x four B fragments (32x16)
        v16h ap, bv0, bv1, bv2, bv3;
        for (int j = 0; j < 8; ++j) {
            int kf = kfrag(j, hi);
            v2h tp = *(const v2h*)&sP[wave][n][kf];
            v2h t0 = *(const v2h*)&sVt[n][kf];
            v2h t1 = *(const v2h*)&sVt[16 + n][kf];
            v2h t2 = *(const v2h*)&sVt[32 + n][kf];
            v2h t3 = *(const v2h*)&sVt[48 + n][kf];
            ap[2 * j]  = tp[0]; ap[2 * j + 1]  = tp[1];
            bv0[2 * j] = t0[0]; bv0[2 * j + 1] = t0[1];
            bv1[2 * j] = t1[0]; bv1[2 * j + 1] = t1[1];
            bv2[2 * j] = t2[0]; bv2[2 * j + 1] = t2[1];
            bv3[2 * j] = t3[0]; bv3[2 * j + 1] = t3[1];
        }
        o0 = __builtin_amdgcn_wmma_f32_16x16x32_f16(false, ap, false, bv0,
                                                    (short)0, o0, false, false);
        o1 = __builtin_amdgcn_wmma_f32_16x16x32_f16(false, ap, false, bv1,
                                                    (short)0, o1, false, false);
        o2 = __builtin_amdgcn_wmma_f32_16x16x32_f16(false, ap, false, bv2,
                                                    (short)0, o2, false, false);
        o3 = __builtin_amdgcn_wmma_f32_16x16x32_f16(false, ap, false, bv3,
                                                    (short)0, o3, false, false);
    }
    // normalize and write attention output in token layout [b][pos][d] (f16)
    for (int r = 0; r < 8; ++r) {
        int m = r + 8 * hi;
        float inv = 1.0f / rsum[r];
        size_t row = (size_t)b * KTOK + row_off + q0 + m;
        _Float16* dst = O + row * DIM + h * HDIM;
        dst[n]      = (_Float16)(o0[r] * inv);
        dst[16 + n] = (_Float16)(o1[r] * inv);
        dst[32 + n] = (_Float16)(o2[r] * inv);
        dst[48 + n] = (_Float16)(o3[r] * inv);
    }
}

// ---------------------------------------------------------------------------
// 6) Output projection GEMM with scatter through `order` into fp32 d_out.
//    Same core as proj_qkv (128x64 block tile, async double buffering).
// ---------------------------------------------------------------------------
__global__ __launch_bounds__(256) void proj_out_kernel(
    const _Float16* __restrict__ A, int row_off,
    const _Float16* __restrict__ W, const float* __restrict__ bias,
    const int* __restrict__ order, float* __restrict__ out) {
    __shared__ _Float16 sA[2][128][KT];
    __shared__ _Float16 sW[2][64][KT];
    const int tid = threadIdx.x;
    const int wave = tid >> 5, lane = tid & 31;
    const int n = lane & 15, hi = lane >> 4;
    const int wm = wave >> 1, wn = wave & 1;
    const int n0 = blockIdx.x * 64;
    const int m0 = blockIdx.y * 128;

    auto stage = [&](int kt, int bf) {
        for (int it = 0; it < 4; ++it) {
            int idx = tid + it * 256;
            int r = idx >> 3, c8 = idx & 7;
            int g = m0 + r, b = g >> 10, m = g & 1023;
            const _Float16* src =
                A + ((size_t)(b * KTOK + row_off + m)) * DIM + kt + 8 * c8;
            async_b128((uint32_t)(size_t)&sA[bf][r][8 * c8], src);
        }
        for (int it = 0; it < 2; ++it) {
            int idx = tid + it * 256;
            int r = idx >> 3, c8 = idx & 7;
            const _Float16* src = W + (size_t)(n0 + r) * DIM + kt + 8 * c8;
            async_b128((uint32_t)(size_t)&sW[bf][r][8 * c8], src);
        }
    };

    v8f c00, c01, c10, c11;
    for (int r = 0; r < 8; ++r) { c00[r] = c01[r] = c10[r] = c11[r] = 0.0f; }

    stage(0, 0);
    int bf = 0;
    for (int kt = 0; kt < DIM; kt += KT) {
        wait_async0();
        __syncthreads();
        if (kt + KT < DIM) stage(kt + KT, bf ^ 1);
        for (int ks = 0; ks < KT; ks += 32) {
            v16h a0, a1, b0, b1;
            for (int j = 0; j < 8; ++j) {
                int kf = ks + kfrag(j, hi);
                v2h t;
                t = *(const v2h*)&sA[bf][wm * 32 + n][kf];
                a0[2 * j] = t[0]; a0[2 * j + 1] = t[1];
                t = *(const v2h*)&sA[bf][wm * 32 + 16 + n][kf];
                a1[2 * j] = t[0]; a1[2 * j + 1] = t[1];
                t = *(const v2h*)&sW[bf][wn * 32 + n][kf];
                b0[2 * j] = t[0]; b0[2 * j + 1] = t[1];
                t = *(const v2h*)&sW[bf][wn * 32 + 16 + n][kf];
                b1[2 * j] = t[0]; b1[2 * j + 1] = t[1];
            }
            c00 = __builtin_amdgcn_wmma_f32_16x16x32_f16(false, a0, false, b0,
                                                         (short)0, c00, false, false);
            c01 = __builtin_amdgcn_wmma_f32_16x16x32_f16(false, a0, false, b1,
                                                         (short)0, c01, false, false);
            c10 = __builtin_amdgcn_wmma_f32_16x16x32_f16(false, a1, false, b0,
                                                         (short)0, c10, false, false);
            c11 = __builtin_amdgcn_wmma_f32_16x16x32_f16(false, a1, false, b1,
                                                         (short)0, c11, false, false);
        }
        __syncthreads();
        bf ^= 1;
    }

    for (int rb = 0; rb < 2; ++rb) {
        for (int cb = 0; cb < 2; ++cb) {
            const v8f& c = rb ? (cb ? c11 : c10) : (cb ? c01 : c00);
            for (int r = 0; r < 8; ++r) {
                int mloc = wm * 32 + rb * 16 + r + 8 * hi;
                int g = m0 + mloc, b = g >> 10, p = g & 1023;
                int orig = order[(size_t)b * KTOK + row_off + p];
                int n_eff = n0 + wn * 32 + cb * 16 + n;
                float val = c[r] + bias[n_eff];
                out[((size_t)b * KTOK + orig) * DIM + n_eff] = val;
            }
        }
    }
}

// ---------------------------------------------------------------------------
// Host launcher
// ---------------------------------------------------------------------------
extern "C" void kernel_launch(void* const* d_in, const int* in_sizes, int n_in,
                              void* d_out, int out_size, void* d_ws,
                              size_t ws_size, hipStream_t stream) {
    const float* x         = (const float*)d_in[0];
    const float* coords    = (const float*)d_in[1];
    const unsigned char* is_ctx = (const unsigned char*)d_in[2];
    const float* rope      = (const float*)d_in[3];
    const float* ctx_in_w  = (const float*)d_in[4];
    const float* ctx_in_b  = (const float*)d_in[5];
    const float* ctx_out_w = (const float*)d_in[6];
    const float* ctx_out_b = (const float*)d_in[7];
    const float* tgt_in_w  = (const float*)d_in[8];
    const float* tgt_in_b  = (const float*)d_in[9];
    const float* tgt_out_w = (const float*)d_in[10];
    const float* tgt_out_b = (const float*)d_in[11];
    float* out = (float*)d_out;

    char* wp = (char*)d_ws;
    auto alloc = [&](size_t bytes) -> char* {
        char* p = wp;
        wp += (bytes + 255) & ~(size_t)255;
        return p;
    };
    int*      order = (int*)alloc((size_t)BATCH * KTOK * sizeof(int));
    _Float16* tok   = (_Float16*)alloc((size_t)BATCH * KTOK * DIM * 2);
    _Float16* w_ci  = (_Float16*)alloc((size_t)3 * DIM * DIM * 2);
    _Float16* w_co  = (_Float16*)alloc((size_t)DIM * DIM * 2);
    _Float16* w_ti  = (_Float16*)alloc((size_t)3 * DIM * DIM * 2);
    _Float16* w_to  = (_Float16*)alloc((size_t)DIM * DIM * 2);
    size_t hb = (size_t)BATCH * NHEAD * NCTX * HDIM * 2;
    _Float16* q_ctx = (_Float16*)alloc(hb);
    _Float16* k_ctx = (_Float16*)alloc(hb);
    _Float16* v_ctx = (_Float16*)alloc(hb);
    _Float16* q_tgt = (_Float16*)alloc(hb);
    _Float16* k_tgt = (_Float16*)alloc(hb);
    _Float16* v_tgt = (_Float16*)alloc(hb);
    _Float16* obuf  = (_Float16*)alloc((size_t)BATCH * KTOK * DIM * 2);

    partition_kernel<<<BATCH, 32, 0, stream>>>(is_ctx, order);
    f32_to_f16_kernel<<<2048, 256, 0, stream>>>(ctx_in_w, w_ci, 3 * DIM * DIM);
    f32_to_f16_kernel<<<1024, 256, 0, stream>>>(ctx_out_w, w_co, DIM * DIM);
    f32_to_f16_kernel<<<2048, 256, 0, stream>>>(tgt_in_w, w_ti, 3 * DIM * DIM);
    f32_to_f16_kernel<<<1024, 256, 0, stream>>>(tgt_out_w, w_to, DIM * DIM);
    rope_gather_kernel<<<BATCH * KTOK, 128, 0, stream>>>(x, coords, rope,
                                                         order, tok);
    // ctx branch: full QKV from ctx tokens with ctx weights
    proj_qkv_kernel<<<dim3(48, 32), 256, 0, stream>>>(
        tok, 0, w_ci, ctx_in_b, 0, q_ctx, k_ctx, v_ctx);
    // tgt branch: Q from tgt tokens; K,V from ctx tokens (tgt weights)
    proj_qkv_kernel<<<dim3(16, 32), 256, 0, stream>>>(
        tok, NCTX, w_ti, tgt_in_b, 0, q_tgt, k_tgt, v_tgt);
    proj_qkv_kernel<<<dim3(32, 32), 256, 0, stream>>>(
        tok, 0, w_ti, tgt_in_b, 1024, q_tgt, k_tgt, v_tgt);
    // attention
    attn_kernel<<<dim3(16, NHEAD, BATCH), 128, 0, stream>>>(
        q_ctx, k_ctx, v_ctx, obuf, 0);
    attn_kernel<<<dim3(16, NHEAD, BATCH), 128, 0, stream>>>(
        q_tgt, k_tgt, v_tgt, obuf, NCTX);
    // output projections + scatter into d_out
    proj_out_kernel<<<dim3(16, 32), 256, 0, stream>>>(
        obuf, 0, w_co, ctx_out_b, order, out);
    proj_out_kernel<<<dim3(16, 32), 256, 0, stream>>>(
        obuf, NCTX, w_to, tgt_out_b, order, out);
}
